// LlamaLayer_14293651161375
// MI455X (gfx1250) — compile-verified, matrix operand order checked by reading, not run on hardware
//
#include <hip/hip_runtime.h>

#define DEVINL __device__ __forceinline__

typedef unsigned short ushort_t;
typedef __attribute__((ext_vector_type(8)))  __bf16 v8bf;
typedef __attribute__((ext_vector_type(16))) __bf16 v16bf;
typedef __attribute__((ext_vector_type(8)))  float  v8f;

static constexpr int Bsz  = 2;
static constexpr int Sq   = 2048;
static constexpr int Hd   = 2048;
static constexpr int NHh  = 16;
static constexpr int HDim = 128;
static constexpr int Iff  = 5632;
static constexpr int Mtok = Bsz * Sq;   // 4096 tokens

// ---------- helpers ----------

DEVINL ushort_t f2bf(float f) {
  union { float f; unsigned u; } v; v.f = f;
  unsigned r = v.u + 0x7FFFu + ((v.u >> 16) & 1u);   // round-to-nearest-even
  return (ushort_t)(r >> 16);
}

// WMMA A/B fragment (16-bit, K contiguous in memory).  Per ISA layout:
// lanes 0-15 hold K {0..7,16..23}, lanes 16-31 hold K {8..15,24..31}.
DEVINL v16bf frag_ld(const ushort_t* rowbase, int lane) {
  const int off = (lane & 16) ? 8 : 0;
  v8bf lo = *reinterpret_cast<const v8bf*>(rowbase + off);
  v8bf hi = *reinterpret_cast<const v8bf*>(rowbase + off + 16);
  return __builtin_shufflevector(lo, hi, 0,1,2,3,4,5,6,7,8,9,10,11,12,13,14,15);
}

DEVINL v8f wmma_bf16(v16bf a, v16bf b, v8f c) {
  return __builtin_amdgcn_wmma_f32_16x16x32_bf16(
      false, a, false, b, (short)0, c, false, false);
}

// CDNA5 async global->LDS DMA (16B per lane), tracked by ASYNCcnt.
DEVINL void async_ld_b128(unsigned lds_off, const ushort_t* g) {
  asm volatile("global_load_async_to_lds_b128 %0, %1, off"
               :: "v"(lds_off), "v"(g) : "memory");
}
DEVINL void wait_async0() {
  asm volatile("s_wait_asynccnt 0x0" ::: "memory");
}
DEVINL unsigned lds_addr32(const void* p) {           // addrspace(3) offset
  return (unsigned)(uintptr_t)p;
}

// ---------- weight convert + transpose: W[k][n] f32 -> Wt[n][k] bf16 ----------
// LDS-tiled 32x32 so both read and write are coalesced.

__global__ __launch_bounds__(256) void wcvt_kernel(
    const float* __restrict__ W, ushort_t* __restrict__ Wt, int K, int N) {
  __shared__ float tile[32][33];
  const int kt = blockIdx.y * 32, nt = blockIdx.x * 32;
  const int tx = threadIdx.x & 31, ty = threadIdx.x >> 5;
#pragma unroll
  for (int r = 0; r < 4; ++r)
    tile[ty + r * 8][tx] = W[(size_t)(kt + ty + r * 8) * N + nt + tx];
  __syncthreads();
#pragma unroll
  for (int r = 0; r < 4; ++r)
    Wt[(size_t)(nt + ty + r * 8) * K + kt + tx] = f2bf(tile[tx][ty + r * 8]);
}

// ---------- RMSNorm (f32 in -> bf16 out), one block per token row ----------

__global__ __launch_bounds__(256) void rmsnorm_kernel(
    const float* __restrict__ x, const float* __restrict__ w,
    ushort_t* __restrict__ out) {
  const int row = blockIdx.x;
  const float* xr = x + (size_t)row * Hd;
  float ss = 0.f;
#pragma unroll
  for (int i = 0; i < 8; ++i) {
    float v = xr[threadIdx.x + i * 256]; ss += v * v;
  }
#pragma unroll
  for (int off = 16; off > 0; off >>= 1) ss += __shfl_xor(ss, off, 32);
  __shared__ float red[8];
  if ((threadIdx.x & 31) == 0) red[threadIdx.x >> 5] = ss;
  __syncthreads();
  float tot = 0.f;
#pragma unroll
  for (int i = 0; i < 8; ++i) tot += red[i];
  const float rs = rsqrtf(tot / (float)Hd + 1e-5f);
#pragma unroll
  for (int i = 0; i < 8; ++i) {
    int c = threadIdx.x + i * 256;
    out[(size_t)row * Hd + c] = f2bf(xr[c] * rs * w[c]);
  }
}

// ---------- bf16 GEMM with double-buffered async global->LDS pipeline --------
// out[M][N] = A[M][K] @ Bt[N][K]^T + bias (+res).  128x128x32 tile, 8 waves,
// each wave a 32x64 subtile (2x4 WMMA accumulators).
// Pipeline: wait(tile t) -> barrier -> issue(t+1, other buf) -> compute(t).
// One barrier per tile; unrolled by 2 + peeled so buf is compile-time const.

template <bool RES, bool OUTBF>
__global__ __launch_bounds__(256) void gemm_bf16_kernel(
    const ushort_t* __restrict__ A, const ushort_t* __restrict__ Bt,
    const float* __restrict__ bias, const float* __restrict__ res,
    void* __restrict__ out, int M, int N, int K) {
  __shared__ __align__(16) ushort_t As[2][128 * 32];
  __shared__ __align__(16) ushort_t Bs[2][128 * 32];
  const int tid = threadIdx.x, lane = tid & 31, wave = tid >> 5;
  const int wm = wave >> 1, wn = wave & 1;
  const int m0 = blockIdx.y * 128, n0 = blockIdx.x * 128;

  v8f acc[2][4];
#pragma unroll
  for (int i = 0; i < 2; ++i)
#pragma unroll
    for (int j = 0; j < 4; ++j) acc[i][j] = 0.f;

  const int lr = tid >> 1, lc = (tid & 1) * 16;
  const ushort_t* gA = A  + (size_t)(m0 + lr) * K + lc;
  const ushort_t* gB = Bt + (size_t)(n0 + lr) * K + lc;
  unsigned ldsA[2], ldsB[2];
#pragma unroll
  for (int s = 0; s < 2; ++s) {
    ldsA[s] = lds_addr32(&As[s][lr * 32 + lc]);
    ldsB[s] = lds_addr32(&Bs[s][lr * 32 + lc]);
  }
  auto issue = [&](int k0, int buf) {
    async_ld_b128(ldsA[buf],      gA + k0);
    async_ld_b128(ldsA[buf] + 16, gA + k0 + 8);
    async_ld_b128(ldsB[buf],      gB + k0);
    async_ld_b128(ldsB[buf] + 16, gB + k0 + 8);
  };
  auto compute = [&](int buf) {
    v16bf af[2], bfr[4];
#pragma unroll
    for (int mi = 0; mi < 2; ++mi)
      af[mi] = frag_ld(&As[buf][(wm * 32 + mi * 16 + (lane & 15)) * 32], lane);
#pragma unroll
    for (int ni = 0; ni < 4; ++ni)
      bfr[ni] = frag_ld(&Bs[buf][(wn * 64 + ni * 16 + (lane & 15)) * 32], lane);
#pragma unroll
    for (int mi = 0; mi < 2; ++mi)
#pragma unroll
      for (int ni = 0; ni < 4; ++ni)
        acc[mi][ni] = wmma_bf16(af[mi], bfr[ni], acc[mi][ni]);
  };

  const int nk = K / 32;   // even (64 or 176)
  issue(0, 0);
  for (int t = 0; t + 2 < nk; t += 2) {
    wait_async0(); __syncthreads();
    issue((t + 1) * 32, 1);
    compute(0);
    wait_async0(); __syncthreads();
    issue((t + 2) * 32, 0);
    compute(1);
  }
  wait_async0(); __syncthreads();
  issue((nk - 1) * 32, 1);
  compute(0);
  wait_async0(); __syncthreads();
  compute(1);

  const int hoff = (lane & 16) ? 8 : 0;
#pragma unroll
  for (int mi = 0; mi < 2; ++mi)
#pragma unroll
    for (int ni = 0; ni < 4; ++ni)
#pragma unroll
      for (int j = 0; j < 8; ++j) {
        const int row = m0 + wm * 32 + mi * 16 + j + hoff;
        const int col = n0 + wn * 64 + ni * 16 + (lane & 15);
        float v = acc[mi][ni][j] + bias[col];
        if constexpr (RES) v += res[(size_t)row * N + col];
        if constexpr (OUTBF)
          ((ushort_t*)out)[(size_t)row * N + col] = f2bf(v);
        else
          ((float*)out)[(size_t)row * N + col] = v;
      }
}

// ---------- fused FC1 + SiLU-gate (async pipeline, dual accumulators) --------

__global__ __launch_bounds__(256) void fc1_silu_kernel(
    const ushort_t* __restrict__ A, const ushort_t* __restrict__ Bt,  // [2I][K]
    const float* __restrict__ bias, ushort_t* __restrict__ act,
    int M, int K) {
  __shared__ __align__(16) ushort_t As[2][128 * 32];
  __shared__ __align__(16) ushort_t Bg[2][128 * 32];
  __shared__ __align__(16) ushort_t Bu[2][128 * 32];
  const int tid = threadIdx.x, lane = tid & 31, wave = tid >> 5;
  const int wm = wave >> 1, wn = wave & 1;
  const int m0 = blockIdx.y * 128, n0 = blockIdx.x * 128;

  v8f accg[2][4], accu[2][4];
#pragma unroll
  for (int i = 0; i < 2; ++i)
#pragma unroll
    for (int j = 0; j < 4; ++j) { accg[i][j] = 0.f; accu[i][j] = 0.f; }

  const int lr = tid >> 1, lc = (tid & 1) * 16;
  const ushort_t* gA = A  + (size_t)(m0 + lr) * K + lc;
  const ushort_t* gG = Bt + (size_t)(n0 + lr) * K + lc;
  const ushort_t* gU = Bt + (size_t)(Iff + n0 + lr) * K + lc;
  unsigned lA[2], lG[2], lU[2];
#pragma unroll
  for (int s = 0; s < 2; ++s) {
    lA[s] = lds_addr32(&As[s][lr * 32 + lc]);
    lG[s] = lds_addr32(&Bg[s][lr * 32 + lc]);
    lU[s] = lds_addr32(&Bu[s][lr * 32 + lc]);
  }
  auto issue = [&](int k0, int buf) {
    async_ld_b128(lA[buf],      gA + k0);
    async_ld_b128(lA[buf] + 16, gA + k0 + 8);
    async_ld_b128(lG[buf],      gG + k0);
    async_ld_b128(lG[buf] + 16, gG + k0 + 8);
    async_ld_b128(lU[buf],      gU + k0);
    async_ld_b128(lU[buf] + 16, gU + k0 + 8);
  };
  auto compute = [&](int buf) {
    v16bf af[2], gf[4], uf[4];
#pragma unroll
    for (int mi = 0; mi < 2; ++mi)
      af[mi] = frag_ld(&As[buf][(wm * 32 + mi * 16 + (lane & 15)) * 32], lane);
#pragma unroll
    for (int ni = 0; ni < 4; ++ni) {
      gf[ni] = frag_ld(&Bg[buf][(wn * 64 + ni * 16 + (lane & 15)) * 32], lane);
      uf[ni] = frag_ld(&Bu[buf][(wn * 64 + ni * 16 + (lane & 15)) * 32], lane);
    }
#pragma unroll
    for (int mi = 0; mi < 2; ++mi)
#pragma unroll
      for (int ni = 0; ni < 4; ++ni) {
        accg[mi][ni] = wmma_bf16(af[mi], gf[ni], accg[mi][ni]);
        accu[mi][ni] = wmma_bf16(af[mi], uf[ni], accu[mi][ni]);
      }
  };

  const int nk = K / 32;
  issue(0, 0);
  for (int t = 0; t + 2 < nk; t += 2) {
    wait_async0(); __syncthreads();
    issue((t + 1) * 32, 1);
    compute(0);
    wait_async0(); __syncthreads();
    issue((t + 2) * 32, 0);
    compute(1);
  }
  wait_async0(); __syncthreads();
  issue((nk - 1) * 32, 1);
  compute(0);
  wait_async0(); __syncthreads();
  compute(1);

  const int hoff = (lane & 16) ? 8 : 0;
#pragma unroll
  for (int mi = 0; mi < 2; ++mi)
#pragma unroll
    for (int ni = 0; ni < 4; ++ni)
#pragma unroll
      for (int j = 0; j < 8; ++j) {
        const int row = m0 + wm * 32 + mi * 16 + j + hoff;
        const int col = n0 + wn * 64 + ni * 16 + (lane & 15);
        float g = accg[mi][ni][j] + bias[col];
        float u = accu[mi][ni][j] + bias[Iff + col];
        float silu = g / (1.f + __expf(-g));
        act[(size_t)row * Iff + col] = f2bf(silu * u);
      }
}

// ---------- flash attention, WMMA for QK^T and PV -----------------------------

__global__ __launch_bounds__(256) void attn_kernel(
    const ushort_t* __restrict__ Q, const ushort_t* __restrict__ Kmat,
    const ushort_t* __restrict__ Vmat, const float* __restrict__ mask,
    ushort_t* __restrict__ Ctx) {
  extern __shared__ ushort_t smem[];
  ushort_t* Ks = smem;                  // [key 128][d 128]
  ushort_t* Vs = smem + 128 * 128;      // [d 128][key 128]
  ushort_t* Ps = smem + 2 * 128 * 128;  // [wave 8][16][128]
  const int tid = threadIdx.x, lane = tid & 31, wave = tid >> 5;
  const int b = blockIdx.z, h = blockIdx.y, q0 = blockIdx.x * 128;
  const size_t base = (size_t)b * Sq * Hd;
  const float scale = 0.08838834764831845f;   // 128^-0.5
  const int hoff = (lane & 16) ? 8 : 0;

  v16bf qf[4];
  {
    const ushort_t* qp =
        Q + base + (size_t)(q0 + wave * 16 + (lane & 15)) * Hd + h * HDim;
#pragma unroll
    for (int ks = 0; ks < 4; ++ks) qf[ks] = frag_ld(qp + ks * 32, lane);
  }

  v8f acc_o[8];
#pragma unroll
  for (int i = 0; i < 8; ++i) acc_o[i] = 0.f;
  float m_r[8], l_r[8];
#pragma unroll
  for (int j = 0; j < 8; ++j) { m_r[j] = -1e30f; l_r[j] = 0.f; }

  const int lr = tid >> 1, lc = (tid & 1) * 64;
  const unsigned ksoff = lds_addr32(&Ks[lr * 128 + lc]);
  for (int t0 = 0; t0 < Sq; t0 += 128) {
    __syncthreads();   // previous tile fully consumed
    // K tile: async global->LDS, row-major [key][d]
    const ushort_t* kg = Kmat + base + (size_t)(t0 + lr) * Hd + h * HDim + lc;
#pragma unroll
    for (int i = 0; i < 8; ++i) async_ld_b128(ksoff + i * 16, kg + i * 8);
    // V tile: vector loads + transpose scatter to [d][key]
    const ushort_t* vg = Vmat + base + (size_t)(t0 + lr) * Hd + h * HDim + lc;
    uint4 vv[8];
#pragma unroll
    for (int i = 0; i < 8; ++i)
      vv[i] = reinterpret_cast<const uint4*>(vg)[i];
#pragma unroll
    for (int i = 0; i < 8; ++i) {
      const int d0 = lc + i * 8;
      const unsigned w0 = vv[i].x, w1 = vv[i].y, w2 = vv[i].z, w3 = vv[i].w;
      Vs[(d0 + 0) * 128 + lr] = (ushort_t)(w0 & 0xffff);
      Vs[(d0 + 1) * 128 + lr] = (ushort_t)(w0 >> 16);
      Vs[(d0 + 2) * 128 + lr] = (ushort_t)(w1 & 0xffff);
      Vs[(d0 + 3) * 128 + lr] = (ushort_t)(w1 >> 16);
      Vs[(d0 + 4) * 128 + lr] = (ushort_t)(w2 & 0xffff);
      Vs[(d0 + 5) * 128 + lr] = (ushort_t)(w2 >> 16);
      Vs[(d0 + 6) * 128 + lr] = (ushort_t)(w3 & 0xffff);
      Vs[(d0 + 7) * 128 + lr] = (ushort_t)(w3 >> 16);
    }
    wait_async0();
    __syncthreads();

    v8f sc[8];
#pragma unroll
    for (int n = 0; n < 8; ++n) sc[n] = 0.f;
#pragma unroll
    for (int n = 0; n < 8; ++n)
#pragma unroll
      for (int ks = 0; ks < 4; ++ks) {
        v16bf kf = frag_ld(&Ks[(n * 16 + (lane & 15)) * 128 + ks * 32], lane);
        sc[n] = wmma_bf16(qf[ks], kf, sc[n]);
      }

#pragma unroll
    for (int n = 0; n < 8; ++n) {
      const int key = t0 + n * 16 + (lane & 15);
      const float mb = (1.0f - mask[b * Sq + key]) * -10000.0f;
#pragma unroll
      for (int j = 0; j < 8; ++j) sc[n][j] = sc[n][j] * scale + mb;
    }

    float mnew[8];
#pragma unroll
    for (int j = 0; j < 8; ++j) {
      float v = sc[0][j];
#pragma unroll
      for (int n = 1; n < 8; ++n) v = fmaxf(v, sc[n][j]);
#pragma unroll
      for (int off = 8; off > 0; off >>= 1) v = fmaxf(v, __shfl_xor(v, off, 32));
      mnew[j] = fmaxf(m_r[j], v);
    }
#pragma unroll
    for (int j = 0; j < 8; ++j) {
      const float f = __expf(m_r[j] - mnew[j]);
      l_r[j] *= f;
#pragma unroll
      for (int n = 0; n < 8; ++n) acc_o[n][j] *= f;
      m_r[j] = mnew[j];
    }
#pragma unroll
    for (int n = 0; n < 8; ++n)
#pragma unroll
      for (int j = 0; j < 8; ++j) sc[n][j] = __expf(sc[n][j] - m_r[j]);
#pragma unroll
    for (int j = 0; j < 8; ++j) {
      float s = 0.f;
#pragma unroll
      for (int n = 0; n < 8; ++n) s += sc[n][j];
#pragma unroll
      for (int off = 8; off > 0; off >>= 1) s += __shfl_xor(s, off, 32);
      l_r[j] += s;
    }

    ushort_t* pw = Ps + wave * (16 * 128);
#pragma unroll
    for (int n = 0; n < 8; ++n)
#pragma unroll
      for (int j = 0; j < 8; ++j)
        pw[(j + hoff) * 128 + n * 16 + (lane & 15)] = f2bf(sc[n][j]);

#pragma unroll
    for (int n2 = 0; n2 < 8; ++n2)
#pragma unroll
      for (int ks2 = 0; ks2 < 4; ++ks2) {
        v16bf pf = frag_ld(&pw[(lane & 15) * 128 + ks2 * 32], lane);
        v16bf vf = frag_ld(&Vs[(n2 * 16 + (lane & 15)) * 128 + ks2 * 32], lane);
        acc_o[n2] = wmma_bf16(pf, vf, acc_o[n2]);
      }
  }

#pragma unroll
  for (int n2 = 0; n2 < 8; ++n2)
#pragma unroll
    for (int j = 0; j < 8; ++j) {
      const int row = q0 + wave * 16 + j + hoff;
      const int d = n2 * 16 + (lane & 15);
      Ctx[base + (size_t)row * Hd + h * HDim + d] = f2bf(acc_o[n2][j] / l_r[j]);
    }
}

// ---------- launch -----------------------------------------------------------

extern "C" void kernel_launch(void* const* d_in, const int* in_sizes, int n_in,
                              void* d_out, int out_size, void* d_ws, size_t ws_size,
                              hipStream_t stream) {
  (void)in_sizes; (void)n_in; (void)out_size; (void)ws_size;
  const float* hs        = (const float*)d_in[0];
  const float* amask     = (const float*)d_in[1];
  const float* q_w       = (const float*)d_in[2];
  const float* q_b       = (const float*)d_in[3];
  const float* k_w       = (const float*)d_in[4];
  const float* k_b       = (const float*)d_in[5];
  const float* v_w       = (const float*)d_in[6];
  const float* v_b       = (const float*)d_in[7];
  const float* o_w       = (const float*)d_in[8];
  const float* o_b       = (const float*)d_in[9];
  const float* fc1_w     = (const float*)d_in[10];
  const float* fc1_b     = (const float*)d_in[11];
  const float* fc2_w     = (const float*)d_in[12];
  const float* fc2_b     = (const float*)d_in[13];
  const float* in_norm_w = (const float*)d_in[14];
  const float* pn_w      = (const float*)d_in[15];

  char* wp = (char*)d_ws;
  auto alloc = [&](size_t bytes) -> char* {
    char* p = wp; wp += (bytes + 255) & ~(size_t)255; return p;
  };
  ushort_t* wq_t  = (ushort_t*)alloc((size_t)Hd * Hd * 2);
  ushort_t* wk_t  = (ushort_t*)alloc((size_t)Hd * Hd * 2);
  ushort_t* wv_t  = (ushort_t*)alloc((size_t)Hd * Hd * 2);
  ushort_t* wo_t  = (ushort_t*)alloc((size_t)Hd * Hd * 2);
  ushort_t* w1_t  = (ushort_t*)alloc((size_t)Hd * 2 * Iff * 2);
  ushort_t* w2_t  = (ushort_t*)alloc((size_t)Iff * Hd * 2);
  ushort_t* xn    = (ushort_t*)alloc((size_t)Mtok * Hd * 2);
  ushort_t* qb    = (ushort_t*)alloc((size_t)Mtok * Hd * 2);
  ushort_t* kb    = (ushort_t*)alloc((size_t)Mtok * Hd * 2);
  ushort_t* vb    = (ushort_t*)alloc((size_t)Mtok * Hd * 2);
  ushort_t* ctxb  = (ushort_t*)alloc((size_t)Mtok * Hd * 2);
  float*    hidden= (float*)   alloc((size_t)Mtok * Hd * 4);
  ushort_t* yn    = (ushort_t*)alloc((size_t)Mtok * Hd * 2);
  ushort_t* actb  = (ushort_t*)alloc((size_t)Mtok * Iff * 2);

  // 1) weight convert + transpose to bf16 [N][K] (coalesced, LDS-tiled)
  wcvt_kernel<<<dim3(Hd / 32, Hd / 32), 256, 0, stream>>>(q_w, wq_t, Hd, Hd);
  wcvt_kernel<<<dim3(Hd / 32, Hd / 32), 256, 0, stream>>>(k_w, wk_t, Hd, Hd);
  wcvt_kernel<<<dim3(Hd / 32, Hd / 32), 256, 0, stream>>>(v_w, wv_t, Hd, Hd);
  wcvt_kernel<<<dim3(Hd / 32, Hd / 32), 256, 0, stream>>>(o_w, wo_t, Hd, Hd);
  wcvt_kernel<<<dim3(2 * Iff / 32, Hd / 32), 256, 0, stream>>>(fc1_w, w1_t, Hd, 2 * Iff);
  wcvt_kernel<<<dim3(Hd / 32, Iff / 32), 256, 0, stream>>>(fc2_w, w2_t, Iff, Hd);

  // 2) input RMSNorm -> bf16
  rmsnorm_kernel<<<Mtok, 256, 0, stream>>>(hs, in_norm_w, xn);

  // 3) QKV projections (bf16 out)
  dim3 gP(Hd / 128, Mtok / 128);
  gemm_bf16_kernel<false, true><<<gP, 256, 0, stream>>>(xn, wq_t, q_b, nullptr, qb, Mtok, Hd, Hd);
  gemm_bf16_kernel<false, true><<<gP, 256, 0, stream>>>(xn, wk_t, k_b, nullptr, kb, Mtok, Hd, Hd);
  gemm_bf16_kernel<false, true><<<gP, 256, 0, stream>>>(xn, wv_t, v_b, nullptr, vb, Mtok, Hd, Hd);

  // 4) flash attention
  attn_kernel<<<dim3(Sq / 128, NHh, Bsz), 256, 3 * 128 * 128 * 2, stream>>>(
      qb, kb, vb, amask, ctxb);

  // 5) output projection + residual -> hidden (f32)
  gemm_bf16_kernel<true, false><<<gP, 256, 0, stream>>>(ctxb, wo_t, o_b, hs, hidden, Mtok, Hd, Hd);

  // 6) post-attention RMSNorm -> bf16
  rmsnorm_kernel<<<Mtok, 256, 0, stream>>>(hidden, pn_w, yn);

  // 7) fused FC1 + SiLU gate -> act (bf16)
  fc1_silu_kernel<<<dim3(Iff / 128, Mtok / 128), 256, 0, stream>>>(
      yn, w1_t, fc1_b, actb, Mtok, Hd);

  // 8) FC2 + residual -> d_out (f32)
  gemm_bf16_kernel<true, false><<<gP, 256, 0, stream>>>(actb, w2_t, fc2_b, hidden,
                                                        (float*)d_out, Mtok, Hd, Iff);
}
